// CausalSelfAttention_62835371540963
// MI455X (gfx1250) — matrix operand order repaired and checked by reading, not compiled, not run on hardware
//
#include <hip/hip_runtime.h>

// ---------------------------------------------------------------------------
// CDNA5 (gfx1250) causal self-attention: bf16 WMMA everywhere, fp32 accum.
// Shapes: B=2, T=2048, C=1024, H=16, D=64.  M = B*T = 4096.
// Flash-attention K/V tiles staged into LDS via the Tensor Data Mover (TDM).
// ---------------------------------------------------------------------------

typedef __attribute__((ext_vector_type(8)))  float        v8f;
typedef __attribute__((ext_vector_type(16))) __bf16       v16bf;
typedef __attribute__((ext_vector_type(8)))  __bf16       v8bf;
typedef __attribute__((ext_vector_type(4)))  unsigned int v4u;
typedef __attribute__((ext_vector_type(8)))  int          v8i;
typedef __attribute__((ext_vector_type(4)))  int          v4i;

union V16 { v16bf v; v8bf h[2]; };

static __device__ __forceinline__ v8f wmma_bf16(v16bf a, v16bf b, v8f c) {
  // D = A(16x32 bf16) * B(32x16 bf16) + C(16x16 f32)
  return __builtin_amdgcn_wmma_f32_16x16x32_bf16(
      /*neg_a=*/false, a, /*neg_b=*/false, b,
      /*c_mod=*/(short)0, c, /*reuse_a=*/false, /*reuse_b=*/false);
}

static __device__ __forceinline__ v8f vzero() {
  v8f z = {0.f,0.f,0.f,0.f,0.f,0.f,0.f,0.f};
  return z;
}

#if defined(__gfx1250__) && __has_builtin(__builtin_amdgcn_tensor_load_to_lds) && \
    __has_builtin(__builtin_amdgcn_s_wait_tensorcnt)
#define USE_TDM 1
#else
#define USE_TDM 0
#endif

#if USE_TDM
// 2D bf16 tile load via Tensor Data Mover: tile (w x h) elems, row stride
// `stride` elems in global, packed w-contiguous rows in LDS at lds_off.
// D# per CDNA5 ISA (groups 0/1; groups 2/3 zero = 2D).
// Toolchain here uses the 6-arg builtin form (extra int32x8 group, then cpol).
static __device__ __forceinline__ void tdm_load_2d(unsigned lds_off, const void* gaddr,
                                                   unsigned w, unsigned h, unsigned stride) {
  unsigned long long ga = (unsigned long long)(uintptr_t)gaddr;
  v4u g0;
  g0.x = 1u;                                                  // count=1, user mode
  g0.y = lds_off;                                             // lds_addr (bytes)
  g0.z = (unsigned)ga;                                        // global_addr[31:0]
  g0.w = (unsigned)((ga >> 32) & 0x01ffffffull) | (2u << 30); // addr[56:32] | type=2
  v8i g1;
  g1[0] = (int)(1u << 16);                                    // data_size=1 -> 2 bytes
  g1[1] = (int)((w & 0xffffu) << 16);                         // tensor_dim0[15:0]
  g1[2] = (int)((w >> 16) | ((h & 0xffffu) << 16));           // tensor_dim0 hi | tensor_dim1 lo
  g1[3] = (int)((h >> 16) | ((w & 0xffffu) << 16));           // tensor_dim1 hi | tile_dim0
  g1[4] = (int)(h & 0xffffu);                                 // tile_dim1 (tile_dim2=0)
  g1[5] = (int)stride;                                        // tensor_dim0_stride[31:0]
  g1[6] = 0;                                                  // stride hi | dim1_stride lo
  g1[7] = 0;
  v4i z4 = {0, 0, 0, 0};
  v8i z8 = {0, 0, 0, 0, 0, 0, 0, 0};
  __builtin_amdgcn_tensor_load_to_lds(g0, g1, z4, z4, z8, 0);
}
#endif

// ---------------------------------------------------------------------------
// fp32 -> bf16 conversion kernels
// ---------------------------------------------------------------------------
__global__ void cvt_bf16(const float* __restrict__ in, __bf16* __restrict__ out, int n) {
  int i = blockIdx.x * blockDim.x + threadIdx.x;
  if (i < n) out[i] = (__bf16)in[i];
}

// in: [rows][cols] fp32 row-major  ->  out: [cols][rows] bf16 (transposed)
__global__ void cvt_transpose_bf16(const float* __restrict__ in, __bf16* __restrict__ out,
                                   int rows, int cols) {
  int i = blockIdx.x * blockDim.x + threadIdx.x;
  if (i >= rows * cols) return;
  int r = i / cols, c = i - r * cols;
  out[(size_t)c * rows + r] = (__bf16)in[(size_t)r * cols + c];
}

// ---------------------------------------------------------------------------
// GEMM 1: qkv = x_bf16[4096x1024] @ W_attnT + b_attn, scattered to
//   q,k : [B,H,T,D] bf16     v : transposed [B,H,D,T] bf16
// One wave per 16x64 output tile.
// ---------------------------------------------------------------------------
__global__ __launch_bounds__(32) void gemm_qkv(
    const __bf16* __restrict__ xb,   // [4096][1024]
    const __bf16* __restrict__ wT,   // [3072][1024]  (N-major, K contiguous)
    const float*  __restrict__ bias, // [3072]
    __bf16* __restrict__ q, __bf16* __restrict__ k, __bf16* __restrict__ vT) {
  const int lane = threadIdx.x;
  const int hi = lane >> 4, lo = lane & 15;
  const int m0 = blockIdx.x * 16;
  const int nbase = blockIdx.y * 64;

  v8f acc[4];
  #pragma unroll
  for (int t = 0; t < 4; ++t) acc[t] = vzero();

  for (int k0 = 0; k0 < 1024; k0 += 32) {
    // A fragment: 16x32 bf16, lane(lo)=row, hi selects K-subgroup
    const __bf16* ap = xb + (size_t)(m0 + lo) * 1024 + k0 + 16 * hi;
    V16 a;
    a.h[0] = *(const v8bf*)ap;
    a.h[1] = *(const v8bf*)(ap + 8);
    if (k0 + 32 < 1024) __builtin_prefetch(ap + 32, 0, 3);
    #pragma unroll
    for (int dt = 0; dt < 4; ++dt) {
      // B fragment: 32x16 bf16, lane(lo)=column n, K contiguous in memory
      const __bf16* bp = wT + (size_t)(nbase + dt * 16 + lo) * 1024 + k0 + 16 * hi;
      V16 bb;
      bb.h[0] = *(const v8bf*)bp;
      bb.h[1] = *(const v8bf*)(bp + 8);
      acc[dt] = wmma_bf16(a.v, bb.v, acc[dt]);
    }
  }

  // epilogue: +bias, convert, scatter into q/k/vT (each 64-col tile stays in one head)
  #pragma unroll
  for (int dt = 0; dt < 4; ++dt) {
    const int n   = nbase + dt * 16 + lo;
    const int sec = n >> 10;        // 0:q 1:k 2:v
    const int cc  = n & 1023;
    const int h   = cc >> 6, d = cc & 63;
    const float bv = bias[n];
    #pragma unroll
    for (int j = 0; j < 8; ++j) {
      const int gm = m0 + j + 8 * hi;          // C layout: row = j + 8*(lane>=16)
      const int b = gm >> 11, t = gm & 2047;
      const __bf16 o = (__bf16)(acc[dt][j] + bv);
      if (sec == 0)      q [(((size_t)(b * 16 + h) * 2048 + t) << 6) + d] = o;
      else if (sec == 1) k [(((size_t)(b * 16 + h) * 2048 + t) << 6) + d] = o;
      else               vT[(((size_t)(b * 16 + h) * 64 + d) << 11) + t] = o;
    }
  }
}

// ---------------------------------------------------------------------------
// Flash attention: one wave per (b*h, 16-query-row block).
// Per 32-key chunk: TDM stages K(32x64) and V^T(64x32) tiles into LDS,
// S = Q K^T via 4 WMMAs, online softmax, P re-laid out through LDS
// (C-fragment -> A-fragment), P@V via 4 WMMAs.
// ---------------------------------------------------------------------------
__global__ __launch_bounds__(32) void flash_attn(
    const __bf16* __restrict__ q,   // [B,H,T,D]
    const __bf16* __restrict__ kk,  // [B,H,T,D]
    const __bf16* __restrict__ vT,  // [B,H,D,T]
    __bf16* __restrict__ y) {       // [B,T,C]
#if USE_TDM
  __shared__ __attribute__((aligned(16))) __bf16 ldsK[32 * 64];  // [key][d]
  __shared__ __attribute__((aligned(16))) __bf16 ldsV[64 * 32];  // [d][key]
#endif
  __shared__ __attribute__((aligned(16))) __bf16 ldsP[16 * 32];

  const int lane = threadIdx.x;
  const int hi = lane >> 4, lo = lane & 15;
  const int bh = blockIdx.x;            // b*16 + h
  const int b = bh >> 4, h = bh & 15;
  const int q0 = blockIdx.y * 16;

  const __bf16* qh = q  + (size_t)bh * 2048 * 64;
  const __bf16* kh = kk + (size_t)bh * 2048 * 64;
  const __bf16* vh = vT + (size_t)bh * 64 * 2048;

  // Preload Q A-fragments for the two 32-wide d-slices
  V16 qa[2];
  #pragma unroll
  for (int s = 0; s < 2; ++s) {
    const __bf16* ap = qh + (size_t)(q0 + lo) * 64 + s * 32 + 16 * hi;
    qa[s].h[0] = *(const v8bf*)ap;
    qa[s].h[1] = *(const v8bf*)(ap + 8);
  }

  float mr[8], lr[8];
  v8f o[4];
  #pragma unroll
  for (int j = 0; j < 8; ++j) { mr[j] = -1e30f; lr[j] = 0.f; }
  #pragma unroll
  for (int t = 0; t < 4; ++t) o[t] = vzero();

  const float scale = 0.125f;           // 1/sqrt(64)
  const int kend = q0 + 15;             // last valid key (causal)

  for (int j0 = 0; j0 <= kend; j0 += 32) {
#if USE_TDM
    // make sure last iteration's ds_loads drained before TDM overwrites LDS
    asm volatile("s_wait_dscnt 0x0" ::: "memory");
    tdm_load_2d((unsigned)(uintptr_t)ldsK, kh + (size_t)j0 * 64, 64u, 32u, 64u);
    tdm_load_2d((unsigned)(uintptr_t)ldsV, vh + j0,              32u, 64u, 2048u);
    __builtin_amdgcn_s_wait_tensorcnt(0);
#endif

    v8f s0 = vzero(), s1 = vzero();     // keys j0..j0+15 and j0+16..j0+31
    #pragma unroll
    for (int s = 0; s < 2; ++s) {
#if USE_TDM
      const __bf16* bp0 = &ldsK[(lo) * 64      + s * 32 + 16 * hi];
      const __bf16* bp1 = &ldsK[(16 + lo) * 64 + s * 32 + 16 * hi];
#else
      const __bf16* bp0 = kh + (size_t)(j0 + lo) * 64      + s * 32 + 16 * hi;
      const __bf16* bp1 = kh + (size_t)(j0 + 16 + lo) * 64 + s * 32 + 16 * hi;
#endif
      V16 b0, b1;
      b0.h[0] = *(const v8bf*)bp0;  b0.h[1] = *(const v8bf*)(bp0 + 8);
      b1.h[0] = *(const v8bf*)bp1;  b1.h[1] = *(const v8bf*)(bp1 + 8);
      s0 = wmma_bf16(qa[s].v, b0.v, s0);
      s1 = wmma_bf16(qa[s].v, b1.v, s1);
    }

    // online softmax: rows span 16 lanes in the C layout; reduce via shfl_xor 1/2/4/8
    float p0[8], p1[8];
    #pragma unroll
    for (int j = 0; j < 8; ++j) {
      const int rq = q0 + j + 8 * hi;                 // this element's query row
      float e0 = s0[j] * scale; if (j0 + lo > rq)      e0 = -1e30f;
      float e1 = s1[j] * scale; if (j0 + 16 + lo > rq) e1 = -1e30f;
      float rm = fmaxf(e0, e1);
      #pragma unroll
      for (int w = 1; w <= 8; w <<= 1) rm = fmaxf(rm, __shfl_xor(rm, w, 32));
      const float mn    = fmaxf(mr[j], rm);
      const float alpha = __expf(mr[j] - mn);
      const float q0e = __expf(e0 - mn);
      const float q1e = __expf(e1 - mn);
      float rs = q0e + q1e;
      #pragma unroll
      for (int w = 1; w <= 8; w <<= 1) rs += __shfl_xor(rs, w, 32);
      lr[j] = lr[j] * alpha + rs;
      mr[j] = mn;
      #pragma unroll
      for (int t = 0; t < 4; ++t) o[t][j] *= alpha;
      p0[j] = q0e; p1[j] = q1e;
    }

    // C-fragment -> row-major 16x32 bf16 in LDS
    #pragma unroll
    for (int j = 0; j < 8; ++j) {
      const int row = j + 8 * hi;
      ldsP[row * 32 + lo]      = (__bf16)p0[j];
      ldsP[row * 32 + lo + 16] = (__bf16)p1[j];
    }
    __builtin_amdgcn_wave_barrier();

    // A-fragment of P from LDS
    V16 pa;
    {
      const __bf16* pp = &ldsP[lo * 32 + 16 * hi];
      pa.h[0] = *(const v8bf*)pp;
      pa.h[1] = *(const v8bf*)(pp + 8);
    }

    // o += P @ V   (V stored key-contiguous: [d][key])
    #pragma unroll
    for (int dt = 0; dt < 4; ++dt) {
#if USE_TDM
      const __bf16* vp = &ldsV[(dt * 16 + lo) * 32 + 16 * hi];
#else
      const __bf16* vp = vh + (size_t)(dt * 16 + lo) * 2048 + j0 + 16 * hi;
#endif
      V16 vb;
      vb.h[0] = *(const v8bf*)vp;
      vb.h[1] = *(const v8bf*)(vp + 8);
      o[dt] = wmma_bf16(pa.v, vb.v, o[dt]);
    }
    __builtin_amdgcn_wave_barrier();
  }

  // normalize and store y[b][t][h*64+d] as bf16
  #pragma unroll
  for (int j = 0; j < 8; ++j) {
    const int row = j + 8 * hi;
    const int t = q0 + row;
    const float inv = 1.0f / lr[j];
    #pragma unroll
    for (int dt = 0; dt < 4; ++dt) {
      const int d = dt * 16 + lo;
      y[((size_t)(b * 2048 + t)) * 1024 + h * 64 + d] = (__bf16)(o[dt][j] * inv);
    }
  }
}

// ---------------------------------------------------------------------------
// GEMM 2: out = y_bf16[4096x1024] @ W_outT + b_out  -> fp32 d_out
// ---------------------------------------------------------------------------
__global__ __launch_bounds__(32) void gemm_proj(
    const __bf16* __restrict__ yb,   // [4096][1024]
    const __bf16* __restrict__ wT,   // [1024][1024] (N-major)
    const float*  __restrict__ bias, // [1024]
    float* __restrict__ out) {       // [4096][1024]
  const int lane = threadIdx.x;
  const int hi = lane >> 4, lo = lane & 15;
  const int m0 = blockIdx.x * 16;
  const int nbase = blockIdx.y * 64;

  v8f acc[4];
  #pragma unroll
  for (int t = 0; t < 4; ++t) acc[t] = vzero();

  for (int k0 = 0; k0 < 1024; k0 += 32) {
    const __bf16* ap = yb + (size_t)(m0 + lo) * 1024 + k0 + 16 * hi;
    V16 a;
    a.h[0] = *(const v8bf*)ap;
    a.h[1] = *(const v8bf*)(ap + 8);
    if (k0 + 32 < 1024) __builtin_prefetch(ap + 32, 0, 3);
    #pragma unroll
    for (int dt = 0; dt < 4; ++dt) {
      const __bf16* bp = wT + (size_t)(nbase + dt * 16 + lo) * 1024 + k0 + 16 * hi;
      V16 bb;
      bb.h[0] = *(const v8bf*)bp;
      bb.h[1] = *(const v8bf*)(bp + 8);
      acc[dt] = wmma_bf16(a.v, bb.v, acc[dt]);
    }
  }

  #pragma unroll
  for (int dt = 0; dt < 4; ++dt) {
    const int n = nbase + dt * 16 + lo;
    const float bv = bias[n];
    #pragma unroll
    for (int j = 0; j < 8; ++j) {
      const int gm = m0 + j + 8 * hi;
      out[(size_t)gm * 1024 + n] = acc[dt][j] + bv;
    }
  }
}

// ---------------------------------------------------------------------------
// Host-side launch
// ---------------------------------------------------------------------------
extern "C" void kernel_launch(void* const* d_in, const int* in_sizes, int n_in,
                              void* d_out, int out_size, void* d_ws, size_t ws_size,
                              hipStream_t stream) {
  const float* x      = (const float*)d_in[0];  // [2,2048,1024]
  const float* W_attn = (const float*)d_in[1];  // [1024,3072]
  const float* b_attn = (const float*)d_in[2];  // [3072]
  const float* W_out  = (const float*)d_in[3];  // [1024,1024]
  const float* b_out  = (const float*)d_in[4];  // [1024]
  float* out = (float*)d_out;

  char* ws = (char*)d_ws;
  // workspace layout (bytes), total ~56 MB
  __bf16* xb  = (__bf16*)(ws + 0);          // x bf16: 16 MB
  __bf16* waT = (__bf16*)(ws + 16777216);   // W_attn^T bf16: 6 MB
  __bf16* woT = (__bf16*)(ws + 23068672);   // W_out^T  bf16: 2 MB
  __bf16* qb  = (__bf16*)(ws + 25165824);   // Q  [B,H,T,D]: 8 MB
  __bf16* kb  = (__bf16*)(ws + 33554432);   // K  [B,H,T,D]: 8 MB
  __bf16* vTb = (__bf16*)(ws + 41943040);   // V^T[B,H,D,T]: 8 MB
  __bf16* yb  = (__bf16*)(ws + 50331648);   // y  [B,T,C]:   8 MB

  cvt_bf16<<<(8388608 + 255) / 256, 256, 0, stream>>>(x, xb, 8388608);
  cvt_transpose_bf16<<<(3145728 + 255) / 256, 256, 0, stream>>>(W_attn, waT, 1024, 3072);
  cvt_transpose_bf16<<<(1048576 + 255) / 256, 256, 0, stream>>>(W_out,  woT, 1024, 1024);

  gemm_qkv<<<dim3(256, 48), 32, 0, stream>>>(xb, waT, b_attn, qb, kb, vTb);
  flash_attn<<<dim3(32, 128), 32, 0, stream>>>(qb, kb, vTb, yb);
  gemm_proj<<<dim3(256, 16), 32, 0, stream>>>(yb, woT, b_out, out);
}
